// KPCN_43714177139033
// MI455X (gfx1250) — matrix operand order repaired and checked by reading, not compiled, not run on hardware
//
#include <hip/hip_runtime.h>
#include <hip/hip_bf16.h>

typedef __bf16 bf16_t;
typedef __attribute__((ext_vector_type(16))) __bf16 v16bf;
typedef __attribute__((ext_vector_type(8)))  __bf16 v8bf;
typedef __attribute__((ext_vector_type(8)))  float   v8f;

#define HW_OUT (92 * 92)
#define RSZ    (8 * 3 * HW_OUT)

// ---------------------------------------------------------------- helpers ---

__global__ void f32_to_bf16_kernel(const float* __restrict__ x,
                                   bf16_t* __restrict__ y, int n) {
    int i = blockIdx.x * blockDim.x + threadIdx.x;
    if (i < n) y[i] = (bf16_t)x[i];
}

// Pack [Cout, Cin, 5, 5] f32 weights into a zero-padded row-major bf16 GEMM A
// of shape [Mpad, Kpad], k = ci*25 + ky*5 + kx (matches flat OIHW order).
__global__ void pack_weights_kernel(const float* __restrict__ w,
                                    bf16_t* __restrict__ A,
                                    int Cout, int Kr, int Kpad, int total) {
    int i = blockIdx.x * blockDim.x + threadIdx.x;
    if (i >= total) return;
    int m = i / Kpad;
    int k = i - m * Kpad;
    float v = 0.0f;
    if (m < Cout && k < Kr) v = w[(size_t)m * Kr + k];
    A[i] = (bf16_t)v;
}

// ------------------------------------------------ implicit-GEMM conv (WMMA) --
// out[b,co,y,x] = sum_{ci,ky,kx} A[co][ci*25+ky*5+kx] * in[b,ci,y+ky,x+kx]
// M = Cout (Mpad mult of 128), N = B*Ho*Wo, K = Cin*25 (Kpad mult of 32).
// Block: 256 threads = 8 waves; 128x64 output tile; BK = 32.
// Wave w: M-subtiles mi0=2*(w>>1), mi0+1; N-subtiles nj0=(w&1)*2, nj0+1.
// => 4 WMMAs / wave / K-step, each fragment reused twice.

#define LDS_STRIDE 40  // bf16 elements per row (80B, 16B aligned, conflict pad)

__global__ __launch_bounds__(256)
void conv5x5_wmma_kernel(const bf16_t* __restrict__ in,
                         const bf16_t* __restrict__ A,
                         const float*  __restrict__ bias,
                         bf16_t* __restrict__ out_bf,
                         float*  __restrict__ out_f32,
                         int Cin, int Cout, int Hin, int Win, int Ho, int Wo,
                         int Kreal, int Kpad, int Ntot, int do_relu) {
    __shared__ __align__(16) bf16_t As[128 * LDS_STRIDE];  // [m][k] row-major
    __shared__ __align__(16) bf16_t Bs[64 * LDS_STRIDE];   // [n][k] row-major

    const int tid    = threadIdx.x;
    const int lane   = tid & 31;
    const int wv     = tid >> 5;
    const int mi0    = (wv >> 1) << 1;   // 0,2,4,6
    const int nj0    = (wv & 1) << 1;    // 0,2
    const int half16 = lane >> 4;        // 0: lanes 0-15, 1: lanes 16-31
    const int l15    = lane & 15;

    const int m0 = blockIdx.y * 128;
    const int n0 = blockIdx.x * 64;

    v8f acc[2][2] = {};

    // --- per-thread B-tile gather coords (fixed n, 8 consecutive k rows) ---
    const int nc    = tid & 63;
    const int kr0   = (tid >> 6) << 3;
    const int n_g   = n0 + nc;
    const int HoWo  = Ho * Wo;
    const int HinWin = Hin * Win;
    const bool nvalid = (n_g < Ntot);
    int base_n = 0;
    if (nvalid) {
        int bb = n_g / HoWo;
        int rem = n_g - bb * HoWo;
        int py = rem / Wo;
        int px = rem - py * Wo;
        base_n = (bb * Cin * Hin + py) * Win + px;
    }
    // incremental (ci, r) state for k = kk + kr0 + j  (k advances 32/iter)
    int ciS[8], rS[8];
    #pragma unroll
    for (int j = 0; j < 8; ++j) {
        int k0 = kr0 + j;                 // <= 31
        int c0 = (k0 >= 25) ? 1 : 0;
        ciS[j] = c0;
        rS[j]  = k0 - c0 * 25;
    }

    // --- per-thread A-tile load coords (128x32 tile, 2 x b128 per thread) ---
    const int arow = tid >> 1;            // 0..127
    const int acg  = (tid & 1) << 4;      // 0 or 16
    const size_t abase = (size_t)(m0 + arow) * (size_t)Kpad + (size_t)acg;

    const int Kfull = Kreal & ~31;        // checked tail handles [Kfull, Kpad)

    for (int kk = 0; kk < Kfull; kk += 32) {
        // stage A tile (128x32 bf16)
        v8bf av0 = *(const v8bf*)(A + abase + kk);
        v8bf av1 = *(const v8bf*)(A + abase + kk + 8);
        *(v8bf*)(&As[arow * LDS_STRIDE + acg])     = av0;
        *(v8bf*)(&As[arow * LDS_STRIDE + acg + 8]) = av1;
        if (kk + 32 < Kpad)
            __builtin_prefetch(A + abase + kk + 32, 0, 3);

        // stage B tile (im2col gather, no bounds checks in main loop)
        v8bf bv;
        if (nvalid) {
            #pragma unroll
            for (int j = 0; j < 8; ++j) {
                int r  = rS[j];
                int ky = (r * 52429) >> 18;     // r/5 for 0 <= r < 25
                int kx = r - ky * 5;
                bv[j] = in[base_n + ciS[j] * HinWin + ky * Win + kx];
            }
        } else {
            #pragma unroll
            for (int j = 0; j < 8; ++j) bv[j] = (bf16_t)0.0f;
        }
        *(v8bf*)(&Bs[nc * LDS_STRIDE + kr0]) = bv;

        // advance (ci, r): k += 32  ->  ci += 1, r += 7, wrap at 25
        #pragma unroll
        for (int j = 0; j < 8; ++j) {
            int r  = rS[j] + 7;
            int ci = ciS[j] + 1;
            if (r >= 25) { r -= 25; ci += 1; }
            rS[j] = r; ciS[j] = ci;
        }

        __syncthreads();

        const v8bf* ra0 = (const v8bf*)(&As[(mi0 * 16 + l15) * LDS_STRIDE]);
        const v8bf* ra1 = (const v8bf*)(&As[((mi0 + 1) * 16 + l15) * LDS_STRIDE]);
        v8bf a0lo = ra0[half16], a0hi = ra0[half16 + 2];
        v8bf a1lo = ra1[half16], a1hi = ra1[half16 + 2];
        v16bf af0, af1;
        #pragma unroll
        for (int j = 0; j < 8; ++j) {
            af0[j] = a0lo[j]; af0[j + 8] = a0hi[j];
            af1[j] = a1lo[j]; af1[j + 8] = a1hi[j];
        }

        const v8bf* rb0 = (const v8bf*)(&Bs[(nj0 * 16 + l15) * LDS_STRIDE]);
        const v8bf* rb1 = (const v8bf*)(&Bs[((nj0 + 1) * 16 + l15) * LDS_STRIDE]);
        v8bf b0lo = rb0[half16], b0hi = rb0[half16 + 2];
        v8bf b1lo = rb1[half16], b1hi = rb1[half16 + 2];
        v16bf bf0, bf1;
        #pragma unroll
        for (int j = 0; j < 8; ++j) {
            bf0[j] = b0lo[j]; bf0[j + 8] = b0hi[j];
            bf1[j] = b1lo[j]; bf1[j + 8] = b1hi[j];
        }

        acc[0][0] = __builtin_amdgcn_wmma_f32_16x16x32_bf16(
                        false, af0, false, bf0, (short)0, acc[0][0], false, false);
        acc[0][1] = __builtin_amdgcn_wmma_f32_16x16x32_bf16(
                        false, af0, false, bf1, (short)0, acc[0][1], false, false);
        acc[1][0] = __builtin_amdgcn_wmma_f32_16x16x32_bf16(
                        false, af1, false, bf0, (short)0, acc[1][0], false, false);
        acc[1][1] = __builtin_amdgcn_wmma_f32_16x16x32_bf16(
                        false, af1, false, bf1, (short)0, acc[1][1], false, false);

        __syncthreads();
    }

    // ------- checked tail iteration (k in [Kfull, Kpad), zero-padded) -------
    if (Kfull < Kpad) {
        const int kk = Kfull;
        v8bf av0 = *(const v8bf*)(A + abase + kk);
        v8bf av1 = *(const v8bf*)(A + abase + kk + 8);
        *(v8bf*)(&As[arow * LDS_STRIDE + acg])     = av0;
        *(v8bf*)(&As[arow * LDS_STRIDE + acg + 8]) = av1;

        v8bf bv;
        #pragma unroll
        for (int j = 0; j < 8; ++j) {
            int k = kk + kr0 + j;
            bf16_t v = (bf16_t)0.0f;
            if (nvalid && k < Kreal) {
                int ci = k / 25;
                int r  = k - ci * 25;
                int ky = (r * 52429) >> 18;
                int kx = r - ky * 5;
                v = in[base_n + ci * HinWin + ky * Win + kx];
            }
            bv[j] = v;
        }
        *(v8bf*)(&Bs[nc * LDS_STRIDE + kr0]) = bv;

        __syncthreads();

        const v8bf* ra0 = (const v8bf*)(&As[(mi0 * 16 + l15) * LDS_STRIDE]);
        const v8bf* ra1 = (const v8bf*)(&As[((mi0 + 1) * 16 + l15) * LDS_STRIDE]);
        v8bf a0lo = ra0[half16], a0hi = ra0[half16 + 2];
        v8bf a1lo = ra1[half16], a1hi = ra1[half16 + 2];
        v16bf af0, af1;
        #pragma unroll
        for (int j = 0; j < 8; ++j) {
            af0[j] = a0lo[j]; af0[j + 8] = a0hi[j];
            af1[j] = a1lo[j]; af1[j + 8] = a1hi[j];
        }

        const v8bf* rb0 = (const v8bf*)(&Bs[(nj0 * 16 + l15) * LDS_STRIDE]);
        const v8bf* rb1 = (const v8bf*)(&Bs[((nj0 + 1) * 16 + l15) * LDS_STRIDE]);
        v8bf b0lo = rb0[half16], b0hi = rb0[half16 + 2];
        v8bf b1lo = rb1[half16], b1hi = rb1[half16 + 2];
        v16bf bf0, bf1;
        #pragma unroll
        for (int j = 0; j < 8; ++j) {
            bf0[j] = b0lo[j]; bf0[j + 8] = b0hi[j];
            bf1[j] = b1lo[j]; bf1[j + 8] = b1hi[j];
        }

        acc[0][0] = __builtin_amdgcn_wmma_f32_16x16x32_bf16(
                        false, af0, false, bf0, (short)0, acc[0][0], false, false);
        acc[0][1] = __builtin_amdgcn_wmma_f32_16x16x32_bf16(
                        false, af0, false, bf1, (short)0, acc[0][1], false, false);
        acc[1][0] = __builtin_amdgcn_wmma_f32_16x16x32_bf16(
                        false, af1, false, bf0, (short)0, acc[1][0], false, false);
        acc[1][1] = __builtin_amdgcn_wmma_f32_16x16x32_bf16(
                        false, af1, false, bf1, (short)0, acc[1][1], false, false);

        __syncthreads();
    }

    // --- epilogue: C/D layout VGPR r -> row mi*16 + half16*8 + r, col l15 ---
    const int nCol[2] = { n0 + nj0 * 16 + l15, n0 + (nj0 + 1) * 16 + l15 };
    int bN[2] = {0, 0}, yN[2] = {0, 0}, xN[2] = {0, 0};
    bool vN[2];
    #pragma unroll
    for (int t = 0; t < 2; ++t) {
        vN[t] = nCol[t] < Ntot;
        if (vN[t]) {
            bN[t] = nCol[t] / HoWo;
            int r = nCol[t] - bN[t] * HoWo;
            yN[t] = r / Wo;
            xN[t] = r - yN[t] * Wo;
        }
    }

    #pragma unroll
    for (int mm = 0; mm < 2; ++mm) {
        const int mbase = m0 + (mi0 + mm) * 16 + half16 * 8;
        #pragma unroll
        for (int r = 0; r < 8; ++r) {
            int co = mbase + r;
            if (co < Cout) {
                float bsv = bias[co];
                #pragma unroll
                for (int t = 0; t < 2; ++t) {
                    if (vN[t]) {
                        float v = acc[mm][t][r] + bsv;
                        if (do_relu) v = fmaxf(v, 0.0f);
                        size_t idx = (((size_t)bN[t] * Cout + co) * Ho + yN[t]) * Wo + xN[t];
                        if (out_bf) out_bf[idx] = (bf16_t)v; else out_f32[idx] = v;
                    }
                }
            }
        }
    }
}

// ----------------------------------------- softmax + 21x21 weighted filter --
__global__ void softmax_filter_kernel(const float* __restrict__ logits,
                                      const float* __restrict__ buf,
                                      float* __restrict__ out) {
    int i = blockIdx.x * blockDim.x + threadIdx.x;
    if (i >= 8 * HW_OUT) return;
    int b = i / HW_OUT;
    int rem = i - b * HW_OUT;
    int y = rem / 92;
    int x = rem - y * 92;

    const float* lg = logits + (size_t)b * 441 * HW_OUT + rem;

    float mx = -3.4e38f;
    for (int p = 0; p < 441; ++p) mx = fmaxf(mx, lg[(size_t)p * HW_OUT]);
    float s = 0.0f;
    for (int p = 0; p < 441; ++p) s += __expf(lg[(size_t)p * HW_OUT] - mx);
    float inv = 1.0f / s;

    float a0 = 0.0f, a1 = 0.0f, a2 = 0.0f;
    for (int p = 0; p < 441; ++p) {
        float w = __expf(lg[(size_t)p * HW_OUT] - mx) * inv;
        int dy = p / 21;
        int dx = p - dy * 21;
        int yy = y + dy - 10;
        int xx = x + dx - 10;
        if ((unsigned)yy < 92u && (unsigned)xx < 92u) {
            size_t base = (((size_t)b * 3) * 128 + (yy + 18)) * 128 + (xx + 18);
            a0 += w * buf[base];
            a1 += w * buf[base + 128 * 128];
            a2 += w * buf[base + 2 * 128 * 128];
        }
    }
    size_t ob = (size_t)b * 3 * HW_OUT + rem;
    out[ob]              = a0;
    out[ob + HW_OUT]     = a1;
    out[ob + 2 * HW_OUT] = a2;
}

// radiance = crop(albedo) * r_diffuse + exp(r_specular) - 1
__global__ void combine_kernel(const float* __restrict__ rdif,
                               const float* __restrict__ rspec,
                               const float* __restrict__ albedo,
                               float* __restrict__ radiance) {
    int i = blockIdx.x * blockDim.x + threadIdx.x;
    if (i >= RSZ) return;
    int bc = i / HW_OUT;            // b*3 + c
    int rem = i - bc * HW_OUT;
    int y = rem / 92;
    int x = rem - y * 92;
    float a = albedo[((size_t)bc * 128 + (y + 18)) * 128 + (x + 18)];
    radiance[i] = a * rdif[i] + expf(rspec[i]) - 1.0f;
}

// -------------------------------------------------------------------- host --

extern "C" void kernel_launch(void* const* d_in, const int* in_sizes, int n_in,
                              void* d_out, int out_size, void* d_ws, size_t ws_size,
                              hipStream_t stream) {
    (void)in_sizes; (void)n_in; (void)out_size; (void)ws_size;

    static const int dims[10] = {27, 100, 100, 100, 100, 100, 100, 100, 100, 441};
    int Hs[10];
    for (int i = 0; i < 10; ++i) Hs[i] = 128 - 4 * i;  // 128 .. 92

    // packed-weight offsets (bf16 elements) for one chain
    size_t woffE[9], wtot = 0;
    int KpadL[9], MpadL[9];
    for (int i = 0; i < 9; ++i) {
        int Kr = dims[i] * 25;
        KpadL[i] = (Kr + 31) & ~31;
        MpadL[i] = (dims[i + 1] + 127) & ~127;
        woffE[i] = wtot;
        wtot += (size_t)MpadL[i] * (size_t)KpadL[i];
    }

    // workspace layout (256B aligned)
    const size_t actElems = (size_t)8 * 100 * 124 * 124;   // max activation
    size_t o = 0;
    auto alignup = [](size_t v) { return (v + 255) & ~(size_t)255; };
    bf16_t* actA   = (bf16_t*)((char*)d_ws + o); o = alignup(o + actElems * sizeof(bf16_t));
    bf16_t* actB   = (bf16_t*)((char*)d_ws + o); o = alignup(o + actElems * sizeof(bf16_t));
    float*  logits = (float*) ((char*)d_ws + o); o = alignup(o + (size_t)8 * 441 * HW_OUT * sizeof(float));
    bf16_t* wpack  = (bf16_t*)((char*)d_ws + o); o = alignup(o + wtot * sizeof(bf16_t));

    float* out0  = (float*)d_out;          // radiance
    float* rdif  = out0 + RSZ;             // r_diffuse
    float* rspec = out0 + 2 * (size_t)RSZ; // r_specular

    for (int c = 0; c < 2; ++c) {
        // input f32 -> bf16
        const int nin = 8 * 27 * 128 * 128;
        f32_to_bf16_kernel<<<(nin + 255) / 256, 256, 0, stream>>>(
            (const float*)d_in[c], actA, nin);

        // pack this chain's weights
        for (int i = 0; i < 9; ++i) {
            int Kr = dims[i] * 25;
            int total = MpadL[i] * KpadL[i];
            pack_weights_kernel<<<(total + 255) / 256, 256, 0, stream>>>(
                (const float*)d_in[5 + c * 18 + 2 * i], wpack + woffE[i],
                dims[i + 1], Kr, KpadL[i], total);
        }

        // conv chain
        bf16_t* cur = actA;
        bf16_t* nxt = actB;
        for (int i = 0; i < 9; ++i) {
            int Cin = dims[i], Cout = dims[i + 1];
            int Hin = Hs[i], Ho = Hs[i + 1];
            int Kr = Cin * 25;
            int Ntot = 8 * Ho * Ho;
            dim3 grid((Ntot + 63) / 64, MpadL[i] / 128);
            conv5x5_wmma_kernel<<<grid, 256, 0, stream>>>(
                cur, wpack + woffE[i], (const float*)d_in[5 + c * 18 + 2 * i + 1],
                (i < 8) ? nxt : nullptr, (i < 8) ? nullptr : logits,
                Cin, Cout, Hin, Hin, Ho, Ho, Kr, KpadL[i], Ntot, (i < 8) ? 1 : 0);
            bf16_t* t = cur; cur = nxt; nxt = t;
        }

        // softmax + weighted filter -> r_diffuse / r_specular sections of d_out
        const int np = 8 * HW_OUT;
        softmax_filter_kernel<<<(np + 255) / 256, 256, 0, stream>>>(
            logits, (const float*)d_in[2 + c], (c == 0) ? rdif : rspec);
    }

    combine_kernel<<<(RSZ + 255) / 256, 256, 0, stream>>>(
        rdif, rspec, (const float*)d_in[4], out0);
}
